// EfficientBiaffineSpanClassifier_42133629173880
// MI455X (gfx1250) — matrix operand order repaired and checked by reading, not compiled, not run on hardware
//
#include <hip/hip_runtime.h>
#include <hip/hip_bf16.h>

// ---------------------------------------------------------------------------
// EfficientBiaffineSpanClassifier for MI455X (gfx1250), wave32 + WMMA bf16.
// B=4, S=1024, I=1024, H=512, O=8.
//
// Pipeline:
//   K0: one-time pack: X->bf16, Wq/Wk/cls -> bf16 transposed (K-contig)
//   K1: qw/kw = RoPE(Xb @ W + b)           -> bf16 workspace [B*S, H]
//   K2: A = 0.5*(qw @ cls[o] + qw/sqrt(H)) -> bf16 workspace [B*O, S, H]
//   K3: logits = mask/tril( A @ kw^T )     -> fp32 d_out     [B*O, S, S]
//
// Every GEMM operand is bf16 with per-lane K-contiguous layout, so each
// fragment is exactly two global_load_b128 per lane; each 32-deep K-step
// issues 8 independent v_wmma per 12 wide loads.
// ---------------------------------------------------------------------------

typedef __attribute__((ext_vector_type(16))) __bf16 v16bf;
typedef __attribute__((ext_vector_type(8)))  __bf16 v8bf;
typedef __attribute__((ext_vector_type(8)))  float  v8f;

#define NEG_BIG 1e12f

__device__ inline v8f wmma_bf16(v16bf a, v16bf b, v8f c) {
  return __builtin_amdgcn_wmma_f32_16x16x32_bf16(false, a, false, b,
                                                 (short)0, c, false, false);
}

__device__ inline v16bf join16(v8bf lo, v8bf hi) {
  v16bf r;
#pragma unroll
  for (int e = 0; e < 8; ++e) { r[e] = lo[e]; r[8 + e] = hi[e]; }
  return r;
}

// A fragment 16x32 bf16, row-major [M, lda]: per-lane 16B at +0 and +32B.
// Layout (CDNA5 ISA 7.12.2): elem e -> K = (e>=8?16:0) + (lane>=16?8:0) + (e&7)
__device__ inline v16bf load_a_bf16(const __bf16* __restrict__ src, int lda,
                                    int mbase, int kbase, int lane) {
  const int m     = mbase + (lane & 15);
  const int khalf = (lane >= 16) ? 8 : 0;
  const __bf16* row = src + (size_t)m * lda + kbase + khalf;
  return join16(*(const v8bf*)(row), *(const v8bf*)(row + 16));
}

// B fragment 32x16 bf16 from K-contiguous (transposed) source:
// B[k][n] = src[n*lds + k]; elem e -> K = kbase + (e>=8?16:0) + (lane>=16?8:0) + (e&7).
// Per-lane 16B at +0 and +32B -> two b128 loads.
__device__ inline v16bf load_bt_bf16(const __bf16* __restrict__ src, int lds,
                                     int kbase, int nbase, int lane) {
  const int n  = nbase + (lane & 15);
  const int k0 = kbase + ((lane >= 16) ? 8 : 0);
  const __bf16* row = src + (size_t)n * lds + k0;
  return join16(*(const v8bf*)(row), *(const v8bf*)(row + 16));
}

// --- K0a: elementwise fp32 -> bf16 ------------------------------------------
__global__ void k0_cvt_bf16(const float* __restrict__ in,
                            __bf16* __restrict__ out, int n) {
  const int i = (blockIdx.x * 256 + threadIdx.x) * 4;
  if (i < n) {
    const float4 v = *(const float4*)(in + i);
    out[i + 0] = (__bf16)v.x;
    out[i + 1] = (__bf16)v.y;
    out[i + 2] = (__bf16)v.z;
    out[i + 3] = (__bf16)v.w;
  }
}

// --- K0b: transpose + convert: out[z][n*K+k] = (bf16) in[z][k*N+n] ----------
// block 256 = 32 x 8, LDS 32x33 tile, coalesced load and store.
__global__ void k0_transpose_cvt(const float* __restrict__ in,
                                 __bf16* __restrict__ out, int K, int N) {
  __shared__ float tile[32][33];
  const size_t zoff = (size_t)blockIdx.z * K * N;
  in  += zoff;
  out += zoff;
  const int nb = blockIdx.x * 32, kb = blockIdx.y * 32;
  const int tx = threadIdx.x & 31, ty = threadIdx.x >> 5;
#pragma unroll
  for (int r = ty; r < 32; r += 8)
    tile[r][tx] = in[(size_t)(kb + r) * N + nb + tx];
  __syncthreads();
#pragma unroll
  for (int r = ty; r < 32; r += 8)
    out[(size_t)(nb + r) * K + kb + tx] = (__bf16)tile[tx][r];
}

// --- K1: projection + bias + RoPE, bf16 in/out ------------------------------
// Wave patch 32x64; block 8 waves as 4(M) x 2(N) -> 128x128 macro-tile.
// grid: (H/128, M1/128, 2)  block: 256
__global__ void k1_proj_rope(const __bf16* __restrict__ Xb,    // [B*S, I]
                             const __bf16* __restrict__ WqT,   // [H, I]
                             const float*  __restrict__ bq,
                             const __bf16* __restrict__ WkT,   // [H, I]
                             const float*  __restrict__ bk,
                             __bf16* __restrict__ qout,        // [B*S, H]
                             __bf16* __restrict__ kout,        // [B*S, H]
                             int N, int K, int S) {
  const int lane = threadIdx.x & 31;
  const int wave = threadIdx.x >> 5;
  const bool is_k = (blockIdx.z != 0);

  const __bf16* WT   = is_k ? WkT : WqT;
  const float*  bias = is_k ? bk : bq;
  __bf16*       out  = is_k ? kout : qout;

  const int mbase = blockIdx.y * 128 + (wave >> 1) * 32;
  const int nbase = blockIdx.x * 128 + (wave & 1) * 64;

  v8f acc[2][4] = {};
  for (int kb = 0; kb < K; kb += 32) {
    v16bf a0 = load_a_bf16(Xb, K, mbase + 0,  kb, lane);
    v16bf a1 = load_a_bf16(Xb, K, mbase + 16, kb, lane);
    v16bf b0 = load_bt_bf16(WT, K, kb, nbase + 0,  lane);
    v16bf b1 = load_bt_bf16(WT, K, kb, nbase + 16, lane);
    v16bf b2 = load_bt_bf16(WT, K, kb, nbase + 32, lane);
    v16bf b3 = load_bt_bf16(WT, K, kb, nbase + 48, lane);
    acc[0][0] = wmma_bf16(a0, b0, acc[0][0]);
    acc[0][1] = wmma_bf16(a0, b1, acc[0][1]);
    acc[0][2] = wmma_bf16(a0, b2, acc[0][2]);
    acc[0][3] = wmma_bf16(a0, b3, acc[0][3]);
    acc[1][0] = wmma_bf16(a1, b0, acc[1][0]);
    acc[1][1] = wmma_bf16(a1, b1, acc[1][1]);
    acc[1][2] = wmma_bf16(a1, b2, acc[1][2]);
    acc[1][3] = wmma_bf16(a1, b3, acc[1][3]);
  }

  // RoPE epilogue. Column h fixed per lane within each sub-tile; rotation
  // partner column h^1 lives in lane^1 (same rows). cos/sin idx = h mod (H/2).
#pragma unroll
  for (int t = 0; t < 4; ++t) {
    const int   h   = nbase + t * 16 + (lane & 15);
    const float bv  = bias[h];
    const int   jj  = h & 255;                     // H/2 = 256
    const float inv = __expf(-(float)jj * 0.035977892078031975f); // ln(1e4)/256
    const float sgn = ((h & 1) == 0) ? -1.f : 1.f;
#pragma unroll
    for (int i = 0; i < 2; ++i) {
#pragma unroll
      for (int r = 0; r < 8; ++r) {
        const int m  = mbase + i * 16 + r + ((lane >= 16) ? 8 : 0);
        float x  = acc[i][t][r] + bv;
        float xp = __shfl_xor(x, 1, 32);           // partner column's value
        const int s = m & (S - 1);
        float sn, c;
        __sincosf((float)s * inv, &sn, &c);
        out[(size_t)m * N + h] = (__bf16)(x * c + sgn * xp * sn);
      }
    }
  }
}

// --- K2: A = 0.5*(qw @ cls[o] + qw * rsqrt(H)) ------------------------------
// Wave patch 32x64; block 8 waves as 4(M) x 2(N) -> 128x128 macro-tile.
// grid: (H/128, S/128, B*O)  block: 256
__global__ void k2_cls_gemm(const __bf16* __restrict__ qw,     // [B*S, H]
                            const __bf16* __restrict__ clsT,   // [O, H(j), H(i)]
                            __bf16* __restrict__ Aout,         // [B*O, S, H]
                            int S, int H, int O) {
  const int lane = threadIdx.x & 31;
  const int wave = threadIdx.x >> 5;
  const int bo   = blockIdx.z;
  const int bb   = bo / O, o = bo % O;

  const __bf16* qbase = qw   + (size_t)bb * S * H;
  const __bf16* WT    = clsT + (size_t)o * H * H;  // WT[j*H+i] = cls[o][i][j]

  const int mbase = blockIdx.y * 128 + (wave >> 1) * 32;
  const int nbase = blockIdx.x * 128 + (wave & 1) * 64;

  v8f acc[2][4] = {};
  for (int kb = 0; kb < H; kb += 32) {
    v16bf a0 = load_a_bf16(qbase, H, mbase + 0,  kb, lane);
    v16bf a1 = load_a_bf16(qbase, H, mbase + 16, kb, lane);
    v16bf b0 = load_bt_bf16(WT, H, kb, nbase + 0,  lane);
    v16bf b1 = load_bt_bf16(WT, H, kb, nbase + 16, lane);
    v16bf b2 = load_bt_bf16(WT, H, kb, nbase + 32, lane);
    v16bf b3 = load_bt_bf16(WT, H, kb, nbase + 48, lane);
    acc[0][0] = wmma_bf16(a0, b0, acc[0][0]);
    acc[0][1] = wmma_bf16(a0, b1, acc[0][1]);
    acc[0][2] = wmma_bf16(a0, b2, acc[0][2]);
    acc[0][3] = wmma_bf16(a0, b3, acc[0][3]);
    acc[1][0] = wmma_bf16(a1, b0, acc[1][0]);
    acc[1][1] = wmma_bf16(a1, b1, acc[1][1]);
    acc[1][2] = wmma_bf16(a1, b2, acc[1][2]);
    acc[1][3] = wmma_bf16(a1, b3, acc[1][3]);
  }

  const float rs = 0.044194173824159216f;          // 1/sqrt(512)
#pragma unroll
  for (int i = 0; i < 2; ++i) {
#pragma unroll
    for (int j = 0; j < 4; ++j) {
      const int n = nbase + j * 16 + (lane & 15);
#pragma unroll
      for (int r = 0; r < 8; ++r) {
        const int m = mbase + i * 16 + r + ((lane >= 16) ? 8 : 0);
        const float qv = (float)qbase[(size_t)m * H + n];
        Aout[((size_t)bo * S + m) * H + n] = (__bf16)(0.5f * (acc[i][j][r] + qv * rs));
      }
    }
  }
}

// --- K3: logits = mask/tril( A @ kw^T ) -------------------------------------
// Wave patch 32x64; block 8 waves as 4(M) x 2(N) -> 128x128 macro-tile.
// grid: (S/128, S/128, B*O)  block: 256
__global__ void k3_logits(const __bf16* __restrict__ Abuf,     // [B*O, S, H]
                          const __bf16* __restrict__ kw,       // [B*S, H]
                          const float*  __restrict__ mask,     // [B, S]
                          float* __restrict__ out,             // [B*O, S, S]
                          int S, int H, int O) {
  const int lane = threadIdx.x & 31;
  const int wave = threadIdx.x >> 5;
  const int bo   = blockIdx.z;
  const int bb   = bo / O;

  const __bf16* Arow = Abuf + (size_t)bo * S * H;
  const __bf16* krow = kw   + (size_t)bb * S * H;

  const int mbase = blockIdx.y * 128 + (wave >> 1) * 32;
  const int nbase = blockIdx.x * 128 + (wave & 1) * 64;

  v8f acc[2][4] = {};
  for (int kb = 0; kb < H; kb += 32) {
    v16bf a0 = load_a_bf16(Arow, H, mbase + 0,  kb, lane);
    v16bf a1 = load_a_bf16(Arow, H, mbase + 16, kb, lane);
    v16bf b0 = load_bt_bf16(krow, H, kb, nbase + 0,  lane);
    v16bf b1 = load_bt_bf16(krow, H, kb, nbase + 16, lane);
    v16bf b2 = load_bt_bf16(krow, H, kb, nbase + 32, lane);
    v16bf b3 = load_bt_bf16(krow, H, kb, nbase + 48, lane);
    acc[0][0] = wmma_bf16(a0, b0, acc[0][0]);
    acc[0][1] = wmma_bf16(a0, b1, acc[0][1]);
    acc[0][2] = wmma_bf16(a0, b2, acc[0][2]);
    acc[0][3] = wmma_bf16(a0, b3, acc[0][3]);
    acc[1][0] = wmma_bf16(a1, b0, acc[1][0]);
    acc[1][1] = wmma_bf16(a1, b1, acc[1][1]);
    acc[1][2] = wmma_bf16(a1, b2, acc[1][2]);
    acc[1][3] = wmma_bf16(a1, b3, acc[1][3]);
  }

#pragma unroll
  for (int j = 0; j < 4; ++j) {
    const int   y    = nbase + j * 16 + (lane & 15);
    const float mval = mask[(size_t)bb * S + y];
    const float moff = (1.f - mval) * NEG_BIG;
#pragma unroll
    for (int i = 0; i < 2; ++i) {
#pragma unroll
      for (int r = 0; r < 8; ++r) {
        const int x = mbase + i * 16 + r + ((lane >= 16) ? 8 : 0);
        float v = acc[i][j][r] * mval - moff;
        if (y < x) v -= NEG_BIG;                   // tril(S,S,-1) mask
        out[((size_t)bo * S + x) * S + y] = v;
      }
    }
  }
}

// ---------------------------------------------------------------------------
extern "C" void kernel_launch(void* const* d_in, const int* in_sizes, int n_in,
                              void* d_out, int out_size, void* d_ws, size_t ws_size,
                              hipStream_t stream) {
  (void)in_sizes; (void)n_in; (void)out_size; (void)ws_size;

  constexpr int B = 4, S = 1024, I = 1024, H = 512, O = 8;

  const float* X   = (const float*)d_in[0];  // [B,S,I]
  const float* msk = (const float*)d_in[1];  // [B,S]
  const float* Wq  = (const float*)d_in[2];  // [I,H]
  const float* bq  = (const float*)d_in[3];  // [H]
  const float* Wk  = (const float*)d_in[4];  // [I,H]
  const float* bk  = (const float*)d_in[5];  // [H]
  const float* cls = (const float*)d_in[6];  // [O,H,H]
  float* out = (float*)d_out;                // [B,O,S,S]

  // workspace (bf16): qw | kw | A | Xb | WqT | WkT | clsT  (~56 MB total)
  __bf16* qb   = (__bf16*)d_ws;
  __bf16* kb   = qb   + (size_t)B * S * H;   //  4 MB
  __bf16* Ab   = kb   + (size_t)B * S * H;   //  8 MB
  __bf16* Xb   = Ab   + (size_t)B * O * S * H; // 40 MB
  __bf16* WqT  = Xb   + (size_t)B * S * I;   // 48 MB
  __bf16* WkT  = WqT  + (size_t)I * H;       // 49 MB
  __bf16* clsT = WkT  + (size_t)I * H;       // 50 MB

  const int M1 = B * S;                      // 4096 rows for projections

  // K0: pack inputs to bf16 (one-time)
  k0_cvt_bf16<<<(B * S * I) / 1024, 256, 0, stream>>>(X, Xb, B * S * I);
  k0_transpose_cvt<<<dim3(H / 32, I / 32, 1), 256, 0, stream>>>(Wq, WqT, I, H);
  k0_transpose_cvt<<<dim3(H / 32, I / 32, 1), 256, 0, stream>>>(Wk, WkT, I, H);
  k0_transpose_cvt<<<dim3(H / 32, H / 32, O), 256, 0, stream>>>(cls, clsT, H, H);

  // K1: macro 128x128, grid (512/128=4, 4096/128=32, 2)
  k1_proj_rope<<<dim3(H / 128, M1 / 128, 2), 256, 0, stream>>>(
      Xb, WqT, bq, WkT, bk, qb, kb, H, I, S);

  // K2: macro 128x128, grid (512/128=4, 1024/128=8, 32)
  k2_cls_gemm<<<dim3(H / 128, S / 128, B * O), 256, 0, stream>>>(
      qb, clsT, Ab, S, H, O);

  // K3: macro 128x128, grid (1024/128=8, 1024/128=8, 32)
  k3_logits<<<dim3(S / 128, S / 128, B * O), 256, 0, stream>>>(
      Ab, kb, msk, out, S, H, O);
}
